// NetworkV2_58162447122939
// MI455X (gfx1250) — matrix operand order repaired and checked by reading, not compiled, not run on hardware
//
#include <hip/hip_runtime.h>
#include <hip/hip_bf16.h>
#include <math.h>

// ---------------------------------------------------------------------------
// MI455X (gfx1250) linear-attention transformer forward.
// GEMMs: LDS double-buffered bf16 WMMA (v_wmma_f32_16x16x32_bf16, fp32 accum).
// Elementwise/LN/32x32 kv state/softmax head: fp32 VALU.
// ---------------------------------------------------------------------------

#define Bq   16
#define Sq   2048
#define Dq   256
#define Hq   8
#define DHq  32
#define FFq  1024
#define DEPTHq 6
#define ACTq 600
#define Mrows (Bq * Sq)   // 32768

typedef __attribute__((ext_vector_type(16))) __bf16 v16bf;
typedef __attribute__((ext_vector_type(8)))  __bf16 v8bf;
typedef __attribute__((ext_vector_type(8)))  float  v8f;

enum { EPI_F32 = 0, EPI_BIAS_RES_F32 = 1, EPI_BIAS_GELU_BF16 = 2 };

// ---------------------------------------------------------------------------
// Pack fp32 [K,N] weight into WMMA B-fragment order, bf16.
// Tile (kt,nt) = 32x16; per tile 32 lanes x 16 contiguous bf16.
// lane element j:  k = kt*32 + (lane/16)*16 + j,  n = nt*16 + (lane%16)
// ---------------------------------------------------------------------------
__global__ void pack_b_kernel(const float* __restrict__ src, __bf16* __restrict__ dst,
                              int K, int N) {
    int idx = blockIdx.x * blockDim.x + threadIdx.x;
    if (idx >= K * N) return;
    int tile   = idx >> 9;
    int within = idx & 511;
    int lane = within >> 4;
    int j    = within & 15;
    int ntiles = N >> 4;
    int kt = tile / ntiles;
    int nt = tile - kt * ntiles;
    int k = (kt << 5) + ((lane >> 4) << 4) + j;
    int n = (nt << 4) + (lane & 15);
    dst[idx] = (__bf16)src[(size_t)k * N + n];
}

// ---------------------------------------------------------------------------
// WMMA GEMM, LDS double-buffered.
//   C[M,N](+epi) = A[M,K](bf16 row-major) x Bpacked[K,N](bf16 fragment order)
// Block: 256 thr = 8 waves. Block tile 128(M) x 128(N), K-step 32.
// Wave (wm=wave>>1, wn=wave&1) owns 32(M) x 64(N) = 8 WMMA accumulators.
// ---------------------------------------------------------------------------
#define ASTRIDE 40   // 128 rows x 40 bf16 (80B row pitch: 16B aligned, bank-spread)

template <int EPI>
__global__ __launch_bounds__(256)
void gemm_bf16_wmma(const __bf16* __restrict__ A, const __bf16* __restrict__ Bp,
                    const float* __restrict__ bias, const float* __restrict__ res,
                    float* __restrict__ Cf, __bf16* __restrict__ Cb,
                    int M, int N, int K) {
    __shared__ __align__(16) __bf16 As[2][128 * ASTRIDE]; // 2 x 10 KB
    __shared__ __align__(16) __bf16 Bs[2][4096];          // 2 x  8 KB

    const int t      = threadIdx.x;
    const int lane   = t & 31;
    const int wave   = t >> 5;
    const int wm     = wave >> 1;           // 0..3
    const int wn     = wave & 1;            // 0..1
    const int mblk   = blockIdx.y * 128;
    const int nblk   = blockIdx.x * 128;
    const int ntiles = N >> 4;
    const int nt0    = nblk >> 4;           // first of 8 n-tiles for this block
    const int KT     = K >> 5;              // K-steps
    const int khalf  = lane >> 4;

    // staging registers (one 32B chunk of A and of B per thread per K-step)
    v8bf ra0, ra1, rb0, rb1;
    const int arow = t >> 1;                // 0..127
    const int acol = (t & 1) << 4;          // 0 or 16

    auto g2r = [&](int kt) {
        const __bf16* agp = A + (size_t)(mblk + arow) * K + (kt << 5) + acol;
        ra0 = *(const v8bf*)(agp);
        ra1 = *(const v8bf*)(agp + 8);
        const __bf16* bgp = Bp + ((size_t)(kt * ntiles + nt0) << 9) + t * 16;
        rb0 = *(const v8bf*)(bgp);
        rb1 = *(const v8bf*)(bgp + 8);
        if (kt + 2 < KT)                      // near-scope speculative prefetch
            __builtin_prefetch(agp + 64, 0, 3);
    };
    auto r2s = [&](int buf) {
        __bf16* as = &As[buf][arow * ASTRIDE + acol];
        *(v8bf*)(as)     = ra0;
        *(v8bf*)(as + 8) = ra1;
        __bf16* bs = &Bs[buf][t * 16];
        *(v8bf*)(bs)     = rb0;
        *(v8bf*)(bs + 8) = rb1;
    };

    v8f acc[2][4];
#pragma unroll
    for (int mt = 0; mt < 2; ++mt)
#pragma unroll
        for (int nt = 0; nt < 4; ++nt) acc[mt][nt] = (v8f){};

    // prologue: stage tile 0
    g2r(0);
    r2s(0);
    __syncthreads();

    for (int kt = 0; kt < KT; ++kt) {
        const int buf = kt & 1;
        if (kt + 1 < KT) g2r(kt + 1);       // overlap global loads with compute

        // A fragments (16x32 each): rows wm*32 + {0..15, 16..31}
        union { v16bf v; v8bf half[2]; } afrag[2];
#pragma unroll
        for (int mt = 0; mt < 2; ++mt) {
            const int ml = wm * 32 + mt * 16 + (lane & 15);
            const __bf16* ap = &As[buf][ml * ASTRIDE + khalf * 16];
            afrag[mt].half[0] = *(const v8bf*)(ap);        // K 0..7   (+khalf*16)
            afrag[mt].half[1] = *(const v8bf*)(ap + 8);    // K 8..15  (+khalf*16)
        }
        // B fragments: n-tiles wn*4 .. wn*4+3 (packed order in LDS)
#pragma unroll
        for (int nt = 0; nt < 4; ++nt) {
            const __bf16* bp = &Bs[buf][(wn * 4 + nt) * 512 + lane * 16];
            union { v16bf v; v8bf half[2]; } bfrag;
            bfrag.half[0] = *(const v8bf*)(bp);
            bfrag.half[1] = *(const v8bf*)(bp + 8);
#pragma unroll
            for (int mt = 0; mt < 2; ++mt)
                acc[mt][nt] = __builtin_amdgcn_wmma_f32_16x16x32_bf16(
                    false, afrag[mt].v, false, bfrag.v, (short)0, acc[mt][nt],
                    false, false);
        }

        if (kt + 1 < KT) {
            r2s(buf ^ 1);                    // waits on its own global loads only
            __syncthreads();
        }
    }

    // C/D layout: lanes 0-15 -> M=r, lanes 16-31 -> M=8+r; N = lane%16.
#pragma unroll
    for (int mt = 0; mt < 2; ++mt) {
        const int rbase = mblk + wm * 32 + mt * 16 + ((lane >> 4) << 3);
#pragma unroll
        for (int nt = 0; nt < 4; ++nt) {
            const int col = nblk + wn * 64 + nt * 16 + (lane & 15);
#pragma unroll
            for (int r = 0; r < 8; ++r) {
                size_t idx = (size_t)(rbase + r) * N + col;
                float v = acc[mt][nt][r];
                if (EPI == EPI_BIAS_RES_F32) {
                    v += bias[col] + res[idx];
                    Cf[idx] = v;
                } else if (EPI == EPI_BIAS_GELU_BF16) {
                    v += bias[col];
                    v = 0.5f * v * (1.0f + erff(v * 0.70710678118f)); // exact gelu
                    Cb[idx] = (__bf16)v;
                } else {
                    Cf[idx] = v;
                }
            }
        }
    }
}

// ---------------------------------------------------------------------------
// h = emb[x] + pos  (one block per (b,s), 256 threads over D)
// ---------------------------------------------------------------------------
__global__ __launch_bounds__(256)
void embed_kernel(const int* __restrict__ x, const float* __restrict__ emb,
                  const float* __restrict__ pos, float* __restrict__ h) {
    int bs = blockIdx.x, t = threadIdx.x;
    int tok = x[bs];
    int s = bs & (Sq - 1);
    h[(size_t)bs * Dq + t] = emb[(size_t)tok * Dq + t] + pos[(size_t)s * Dq + t];
}

// ---------------------------------------------------------------------------
// LayerNorm over D=256, one row per block. OUT_BF=1 -> bf16 (GEMM A operand).
// ---------------------------------------------------------------------------
template <int OUT_BF>
__global__ __launch_bounds__(256)
void ln_kernel(const float* __restrict__ x, const float* __restrict__ g,
               const float* __restrict__ bta, __bf16* __restrict__ ob,
               float* __restrict__ of) {
    __shared__ float red[256];
    __shared__ float smean, svar;
    int row = blockIdx.x, t = threadIdx.x;
    float v = x[(size_t)row * Dq + t];
    red[t] = v; __syncthreads();
    for (int o = 128; o > 0; o >>= 1) { if (t < o) red[t] += red[t + o]; __syncthreads(); }
    if (t == 0) smean = red[0] * (1.0f / Dq);
    __syncthreads();
    float d = v - smean;
    red[t] = d * d; __syncthreads();
    for (int o = 128; o > 0; o >>= 1) { if (t < o) red[t] += red[t + o]; __syncthreads(); }
    if (t == 0) svar = red[0] * (1.0f / Dq);
    __syncthreads();
    float y = d * rsqrtf(svar + 1e-5f) * g[t] + bta[t];
    if (OUT_BF) ob[(size_t)row * Dq + t] = (__bf16)y;
    else        of[(size_t)row * Dq + t] = y;
}

// ---------------------------------------------------------------------------
// qf = elu(q)+1 ; kf = (elu(k)+1)*mask ; vf = v*mask  -> [B,H,S,DH] fp32
// idx bits: d[0:4] s[5:15] h[16:18] b[19:22]; write index == idx.
// ---------------------------------------------------------------------------
__global__ __launch_bounds__(256)
void featurize_kernel(const float* __restrict__ qkv, const bool* __restrict__ mask,
                      float* __restrict__ qf, float* __restrict__ kf,
                      float* __restrict__ vf) {
    int idx = blockIdx.x * 256 + threadIdx.x;
    int d = idx & 31;
    int s = (idx >> 5) & (Sq - 1);
    int hh = (idx >> 16) & 7;
    int b = idx >> 19;
    size_t src = ((size_t)(b * Sq + s)) * (3 * Dq) + hh * DHq + d;
    float q = qkv[src], k = qkv[src + Dq], v = qkv[src + 2 * Dq];
    float mf = mask[b * Sq + s] ? 1.0f : 0.0f;
    float qe = (q > 0.0f) ? (q + 1.0f) : __expf(q);
    float ke = (k > 0.0f) ? (k + 1.0f) : __expf(k);
    qf[idx] = qe;
    kf[idx] = ke * mf;
    vf[idx] = v * mf;
}

// ---------------------------------------------------------------------------
// kv[d,e] = sum_s kf[s,d]*vf[s,e];  ksum[d] = sum_s kf[s,d].  One block/(b,h).
// ---------------------------------------------------------------------------
__global__ __launch_bounds__(1024)
void kv_kernel(const float* __restrict__ kf, const float* __restrict__ vf,
               float* __restrict__ kv, float* __restrict__ ksum) {
    int bh = blockIdx.x, t = threadIdx.x;
    int d = t >> 5, e = t & 31;
    const float* kb = kf + (size_t)bh * Sq * DHq;
    const float* vb = vf + (size_t)bh * Sq * DHq;
    float acc = 0.0f, ks = 0.0f;
    for (int s = 0; s < Sq; ++s) {
        float kk = kb[s * DHq + d];
        acc += kk * vb[s * DHq + e];
        ks  += kk;
    }
    kv[(size_t)bh * 1024 + t] = acc;
    if (e == 0) ksum[bh * DHq + d] = ks;
}

// ---------------------------------------------------------------------------
// out[s,e] = (sum_d qf[s,d]*kv[d,e]) / (sum_d qf[s,d]*ksum[d] + eps)
// Writes bf16 into attn[B*S, 256] as the next GEMM's A operand.
// ---------------------------------------------------------------------------
__global__ __launch_bounds__(256)
void attn_out_kernel(const float* __restrict__ qf, const float* __restrict__ kv,
                     const float* __restrict__ ksum, __bf16* __restrict__ attn) {
    __shared__ float skv[1024];
    __shared__ float sks[32];
    int bh = blockIdx.x, t = threadIdx.x;
    for (int i = 0; i < 4; ++i) skv[t + i * 256] = kv[(size_t)bh * 1024 + t + i * 256];
    if (t < 32) sks[t] = ksum[bh * DHq + t];
    __syncthreads();
    int srow = blockIdx.y * 8 + (t >> 5);
    int e = t & 31;
    const float* qrow = qf + (size_t)bh * Sq * DHq + (size_t)srow * DHq;
    float acc = 0.0f, zdot = 0.0f;
    for (int d = 0; d < DHq; ++d) {
        float qv = qrow[d];
        acc  += qv * skv[d * 32 + e];
        zdot += qv * sks[d];
    }
    float outv = acc / (zdot + 1e-6f);
    int b = bh >> 3, hh = bh & 7;
    attn[((size_t)(b * Sq + srow)) * Dq + hh * DHq + e] = (__bf16)outv;
}

// ---------------------------------------------------------------------------
// Head: masked mean-pool -> policy softmax (600) + value tanh. One block per b.
// ---------------------------------------------------------------------------
__global__ __launch_bounds__(256)
void head_kernel(const float* __restrict__ hln, const bool* __restrict__ mask,
                 const float* __restrict__ Wp, const float* __restrict__ bp,
                 const float* __restrict__ Wv1, const float* __restrict__ bv1,
                 const float* __restrict__ Wv2, const float* __restrict__ bv2,
                 float* __restrict__ out) {
    __shared__ float pooled[256];
    __shared__ float logits[ACTq];
    __shared__ float red[256];
    __shared__ float hv[128];
    __shared__ float smax, ssum;
    int b = blockIdx.x, t = threadIdx.x;
    const float* hb = hln + (size_t)b * Sq * Dq;
    const bool*  mb = mask + b * Sq;
    float sum = 0.0f, cnt = 0.0f;
    for (int s = 0; s < Sq; ++s) {
        float mf = mb[s] ? 1.0f : 0.0f;
        sum += hb[(size_t)s * Dq + t] * mf;
        cnt += mf;
    }
    pooled[t] = sum / fmaxf(cnt, 1.0f);
    __syncthreads();
    for (int j = t; j < ACTq; j += 256) {
        float acc = bp[j];
        for (int d = 0; d < Dq; ++d) acc += pooled[d] * Wp[(size_t)d * ACTq + j];
        logits[j] = acc;
    }
    __syncthreads();
    float mx = -1e30f;
    for (int j = t; j < ACTq; j += 256) mx = fmaxf(mx, logits[j]);
    red[t] = mx; __syncthreads();
    for (int o = 128; o > 0; o >>= 1) { if (t < o) red[t] = fmaxf(red[t], red[t + o]); __syncthreads(); }
    if (t == 0) smax = red[0];
    __syncthreads();
    float es = 0.0f;
    for (int j = t; j < ACTq; j += 256) es += __expf(logits[j] - smax);
    red[t] = es; __syncthreads();
    for (int o = 128; o > 0; o >>= 1) { if (t < o) red[t] += red[t + o]; __syncthreads(); }
    if (t == 0) ssum = red[0];
    __syncthreads();
    for (int j = t; j < ACTq; j += 256)
        out[b * ACTq + j] = __expf(logits[j] - smax) / ssum;
    if (t < 128) {
        float acc = bv1[t];
        for (int d = 0; d < Dq; ++d) acc += pooled[d] * Wv1[(size_t)d * 128 + t];
        hv[t] = fmaxf(acc, 0.0f);
    }
    __syncthreads();
    if (t == 0) {
        float acc = bv2[0];
        for (int j = 0; j < 128; ++j) acc += hv[j] * Wv2[j];
        out[Bq * ACTq + b] = tanhf(acc);
    }
}

// ---------------------------------------------------------------------------
extern "C" void kernel_launch(void* const* d_in, const int* in_sizes, int n_in,
                              void* d_out, int out_size, void* d_ws, size_t ws_size,
                              hipStream_t stream) {
    (void)in_sizes; (void)n_in; (void)out_size; (void)ws_size;
    const int*   x     = (const int*)d_in[0];
    const bool*  mask  = (const bool*)d_in[1];
    const float* emb   = (const float*)d_in[2];
    const float* pos   = (const float*)d_in[3];
    const float* ln1_g = (const float*)d_in[4];
    const float* ln1_b = (const float*)d_in[5];
    const float* Wqkv  = (const float*)d_in[6];
    const float* Wout  = (const float*)d_in[7];
    const float* bout  = (const float*)d_in[8];
    const float* ln2_g = (const float*)d_in[9];
    const float* ln2_b = (const float*)d_in[10];
    const float* Wff1  = (const float*)d_in[11];
    const float* bff1  = (const float*)d_in[12];
    const float* Wff2  = (const float*)d_in[13];
    const float* bff2  = (const float*)d_in[14];
    const float* lnf_g = (const float*)d_in[15];
    const float* lnf_b = (const float*)d_in[16];
    const float* Wp    = (const float*)d_in[17];
    const float* bp    = (const float*)d_in[18];
    const float* Wv1   = (const float*)d_in[19];
    const float* bv1   = (const float*)d_in[20];
    const float* Wv2   = (const float*)d_in[21];
    const float* bv2   = (const float*)d_in[22];
    float* out = (float*)d_out;

    // ---- workspace carve (256B aligned) ----
    char* w = (char*)d_ws;
    auto carve = [&](size_t bytes) -> void* {
        void* p = (void*)w;
        w += (bytes + 255) & ~(size_t)255;
        return p;
    };
    float*  h       = (float*) carve((size_t)Mrows * Dq * 4);          // 32 MB
    __bf16* hn_bf   = (__bf16*)carve((size_t)Mrows * Dq * 2);          // 16 MB
    float*  qkv     = (float*) carve((size_t)Mrows * 3 * Dq * 4);      // 96 MB (aliased below)
    float*  qf      = (float*) carve((size_t)Bq * Hq * Sq * DHq * 4);  // 32 MB
    float*  kf      = (float*) carve((size_t)Bq * Hq * Sq * DHq * 4);
    float*  vf      = (float*) carve((size_t)Bq * Hq * Sq * DHq * 4);
    float*  kv      = (float*) carve((size_t)Bq * Hq * 1024 * 4);
    float*  ksum    = (float*) carve((size_t)Bq * Hq * DHq * 4);
    __bf16* attn_bf = (__bf16*)carve((size_t)Mrows * Dq * 2);          // 16 MB
    __bf16* pWqkv   = (__bf16*)carve((size_t)DEPTHq * Dq * 3 * Dq * 2);
    __bf16* pWout   = (__bf16*)carve((size_t)DEPTHq * Dq * Dq * 2);
    __bf16* pWff1   = (__bf16*)carve((size_t)DEPTHq * Dq * FFq * 2);
    __bf16* pWff2   = (__bf16*)carve((size_t)DEPTHq * FFq * Dq * 2);
    // aliases into the qkv region (qkv is dead by the time these are live):
    __bf16* ff1_bf  = (__bf16*)qkv;   // 64 MB <= 96 MB
    float*  hln     = qkv;            // 32 MB <= 96 MB (final LN output)

    // ---- pack weights to WMMA-fragment bf16 layout ----
    for (int i = 0; i < DEPTHq; ++i) {
        { int K = Dq, N = 3 * Dq;
          pack_b_kernel<<<(K * N + 255) / 256, 256, 0, stream>>>(Wqkv + (size_t)i * K * N, pWqkv + (size_t)i * K * N, K, N); }
        { int K = Dq, N = Dq;
          pack_b_kernel<<<(K * N + 255) / 256, 256, 0, stream>>>(Wout + (size_t)i * K * N, pWout + (size_t)i * K * N, K, N); }
        { int K = Dq, N = FFq;
          pack_b_kernel<<<(K * N + 255) / 256, 256, 0, stream>>>(Wff1 + (size_t)i * K * N, pWff1 + (size_t)i * K * N, K, N); }
        { int K = FFq, N = Dq;
          pack_b_kernel<<<(K * N + 255) / 256, 256, 0, stream>>>(Wff2 + (size_t)i * K * N, pWff2 + (size_t)i * K * N, K, N); }
    }

    // ---- embed ----
    embed_kernel<<<Mrows, 256, 0, stream>>>(x, emb, pos, h);

    // ---- layers ----
    for (int i = 0; i < DEPTHq; ++i) {
        // LN1 -> bf16
        ln_kernel<1><<<Mrows, 256, 0, stream>>>(h, ln1_g + i * Dq, ln1_b + i * Dq, hn_bf, nullptr);
        // qkv = hn @ Wqkv   [32768,256]x[256,768]
        gemm_bf16_wmma<EPI_F32><<<dim3(3 * Dq / 128, Mrows / 128), 256, 0, stream>>>(
            hn_bf, pWqkv + (size_t)i * Dq * 3 * Dq, nullptr, nullptr, qkv, nullptr,
            Mrows, 3 * Dq, Dq);
        // elu+1 features
        featurize_kernel<<<(Bq * Hq * Sq * DHq) / 256, 256, 0, stream>>>(qkv, mask, qf, kf, vf);
        // kv state + ksum
        kv_kernel<<<Bq * Hq, 1024, 0, stream>>>(kf, vf, kv, ksum);
        // normalized attention output -> bf16 [32768,256]
        attn_out_kernel<<<dim3(Bq * Hq, Sq / 8), 256, 0, stream>>>(qf, kv, ksum, attn_bf);
        // h = h + attn @ Wout + bout   (in-place residual)
        gemm_bf16_wmma<EPI_BIAS_RES_F32><<<dim3(Dq / 128, Mrows / 128), 256, 0, stream>>>(
            attn_bf, pWout + (size_t)i * Dq * Dq, bout + i * Dq, h, h, nullptr,
            Mrows, Dq, Dq);
        // LN2 -> bf16
        ln_kernel<1><<<Mrows, 256, 0, stream>>>(h, ln2_g + i * Dq, ln2_b + i * Dq, hn_bf, nullptr);
        // ff1 = gelu(hn @ Wff1 + bff1) -> bf16
        gemm_bf16_wmma<EPI_BIAS_GELU_BF16><<<dim3(FFq / 128, Mrows / 128), 256, 0, stream>>>(
            hn_bf, pWff1 + (size_t)i * Dq * FFq, bff1 + i * FFq, nullptr, nullptr, ff1_bf,
            Mrows, FFq, Dq);
        // h = h + ff1 @ Wff2 + bff2
        gemm_bf16_wmma<EPI_BIAS_RES_F32><<<dim3(Dq / 128, Mrows / 128), 256, 0, stream>>>(
            ff1_bf, pWff2 + (size_t)i * FFq * Dq, bff2 + i * Dq, h, h, nullptr,
            Mrows, Dq, FFq);
    }

    // ---- final LN (fp32) + head ----
    ln_kernel<0><<<Mrows, 256, 0, stream>>>(h, lnf_g, lnf_b, nullptr, hln);
    head_kernel<<<Bq, 256, 0, stream>>>(hln, mask, Wp, bp, Wv1, bv1, Wv2, bv2, out);
}
